// DenoiseModule_12489764897172
// MI455X (gfx1250) — matrix operand order repaired and checked by reading, not compiled, not run on hardware
//
#include <hip/hip_runtime.h>
#include <hip/hip_bf16.h>
#include <math.h>

typedef _Float16 f16;
typedef __attribute__((ext_vector_type(16))) _Float16 v16h;
typedef __attribute__((ext_vector_type(8)))  _Float16 v8h;
typedef __attribute__((ext_vector_type(8)))  float    v8f;

#define NB 32
#define NC 3
#define NH 512
#define NW 512
#define NROW   (NB*NC*NH)        // 49152 independent 512-pt transforms
#define TROWS  16                // WMMA M dimension per tile
#define NTILES (NROW/TROWS)      // 3072
#define PLANE  (NH*NW)           // 262144
#define NPIX   ((size_t)NB*NC*NH*NW)

// ---- workspace byte offsets (~103 MB total) ----
#define OFF_FRE   (size_t)0
#define OFF_FIM   (OFF_FRE + (size_t)512*512*2)
#define OFF_GRE   (OFF_FIM + (size_t)512*512*2)
#define OFF_GIM   (OFF_GRE + (size_t)512*512*2)
#define OFF_MW    (OFF_GIM + (size_t)512*512*2)
#define OFF_MH    (OFF_MW  + 512*4)
#define OFF_NOISE (OFF_MH  + 512*4)
#define OFF_Y     ((OFF_NOISE + 96*4 + 255) & ~(size_t)255)

// ------------------------------------------------------------------
// Kernel 0: DFT / IDFT f16 tables (symmetric, so B-operand order is
// just [col*512 + k]), separable shifted-Gaussian mask tables, and
// zeroed per-(b,c) noise accumulators.
// ------------------------------------------------------------------
__global__ __launch_bounds__(256)
void init_tables(f16* __restrict__ Fre, f16* __restrict__ Fim,
                 f16* __restrict__ Gre, f16* __restrict__ Gim,
                 float* __restrict__ mw, float* __restrict__ mh,
                 float* __restrict__ noiseAcc) {
    int idx = blockIdx.x * blockDim.x + threadIdx.x;   // 0..262143
    int n = idx >> 9, k = idx & 511;
    int t = (n * k) & 511;                             // exact mod-512 phase
    float ang = (float)t * (6.28318530717958647692f / 512.0f);
    float s, c;
    __sincosf(ang, &s, &c);
    Fre[idx] = (f16)c;                                 // exp(-i2πnk/512)
    Fim[idx] = (f16)(-s);
    Gre[idx] = (f16)(c * (1.0f / 512.0f));             // exp(+i2πnk/512)/512
    Gim[idx] = (f16)(s * (1.0f / 512.0f));
    if (n == 0) {
        // shifted-mask axis value at index k: linspace(-256,256,512)
        float x = -256.0f + (float)((k + 256) & 511) * (512.0f / 511.0f);
        float v = __expf(-(x * x) * (1.0f / 3200.0f)); // 2*40^2 = 3200
        mw[k] = v;
        mh[k] = v;
    }
    if (idx < 96) noiseAcc[idx] = 0.0f;
}

// ------------------------------------------------------------------
// Kernel 1: 16 rows/block. Input staged via CDNA5 async global->LDS
// DMA (ASYNCcnt), forward DFT (WMMA f16->f32), separable mask in f32,
// inverse DFT (complex x complex = 4 WMMA streams), magnitude out.
// ------------------------------------------------------------------
__device__ __forceinline__ v16h ld_a16(const f16* p) {     // 2x16B -> A/B operand
    v8h lo = *(const v8h*)p;
    v8h hi = *(const v8h*)(p + 16);
    return __builtin_shufflevector(lo, hi, 0,1,2,3,4,5,6,7,8,9,10,11,12,13,14,15);
}

__global__ __launch_bounds__(128)
void fft_filter_kernel(const float* __restrict__ img,
                       const f16* __restrict__ Fre, const f16* __restrict__ Fim,
                       const f16* __restrict__ Gre, const f16* __restrict__ Gim,
                       const float* __restrict__ mw, const float* __restrict__ mh,
                       float* __restrict__ y) {
    // 48 KB: Xs f16 [0,16K) | Sre f16 [16K,32K) | Sim f16 [32K,48K)
    // f32 staging buffer (32 KB) aliases Sre+Sim (dead until forward phase).
    __shared__ __align__(16) unsigned char smem[3 * TROWS * 512 * 2];
    __shared__ float mh16[TROWS];
    f16*   Xs    = (f16*)smem;
    f16*   Sre   = (f16*)(smem + 16384);
    f16*   Sim   = (f16*)(smem + 32768);
    float* stage = (float*)(smem + 16384);

    const int tid = threadIdx.x;
    const int r0  = blockIdx.x * TROWS;

    // ---- async DMA: 16 rows of f32 (32 KB) directly into LDS ----
    {
        const unsigned lds_base = (unsigned)(uintptr_t)stage;  // AS(3) offset in low bits
        const float* gsrc = img + (size_t)r0 * 512;
        #pragma unroll
        for (int it = 0; it < 16; ++it) {                      // 128 thr * 16B * 16 = 32 KB
            const int i = 4 * tid + 512 * it;                  // f32 element index
            unsigned lds_addr = lds_base + (unsigned)(i * 4);
            unsigned long long gaddr = (unsigned long long)(uintptr_t)(gsrc + i);
            asm volatile("global_load_async_to_lds_b128 %0, %1, off"
                         :: "v"(lds_addr), "v"(gaddr) : "memory");
        }
        asm volatile("s_wait_asynccnt 0x0" ::: "memory");
    }
    if (tid < TROWS) mh16[tid] = mh[(r0 + tid) & 511];
    __syncthreads();                      // staged data visible to all waves

    // convert staged f32 -> f16 A-tile (x255/255 cancel through linear ops + abs)
    for (int i = tid; i < TROWS * 512; i += 128)
        Xs[i] = (f16)stage[i];
    __syncthreads();                      // last read of staging before Sre/Sim reuse

    const int wv   = tid >> 5;
    const int lane = tid & 31;
    const int lh   = lane & 15;
    const int kbA  = (lane < 16) ? 0 : 8;   // ISA 16-bit A-matrix layout
    const int kbB  = (lane < 16) ? 0 : 8;   // ISA 16-bit B-matrix layout
    const int mO   = (lane < 16) ? 0 : 8;   // C/D row base per lane half

    // ---------- forward: S = X * (Fre + i*Fim), X real ----------
    for (int j = wv; j < 32; j += 4) {
        const int col = 16 * j + lh;        // frequency bin for this lane
        v8f accR = {}; v8f accI = {};
        for (int kc = 0; kc < 16; ++kc) {
            v16h a   = ld_a16(Xs + lh * 512 + 32 * kc + kbA);
            v16h bre = ld_a16(Fre + (size_t)col * 512 + 32 * kc + kbB);
            v16h bim = ld_a16(Fim + (size_t)col * 512 + 32 * kc + kbB);
            accR = __builtin_amdgcn_wmma_f32_16x16x32_f16(false, a, false, bre,
                                                          (short)0, accR, false, false);
            accI = __builtin_amdgcn_wmma_f32_16x16x32_f16(false, a, false, bim,
                                                          (short)0, accI, false, false);
        }
        const float mwk = mw[col];
        #pragma unroll
        for (int i = 0; i < 8; ++i) {
            const int Mi = i + mO;
            const float mval = mwk * mh16[Mi];
            Sre[Mi * 512 + col] = (f16)(accR[i] * mval);
            Sim[Mi * 512 + col] = (f16)(accI[i] * mval);
        }
    }
    __syncthreads();

    // ---------- inverse: y = S * (Gre + i*Gim), take |y| ----------
    for (int j = wv; j < 32; j += 4) {
        const int col = 16 * j + lh;        // output sample w
        v8f aRR = {}, aII = {}, aRI = {}, aIR = {};
        for (int kc = 0; kc < 16; ++kc) {
            v16h aRe = ld_a16(Sre + lh * 512 + 32 * kc + kbA);
            v16h aIm = ld_a16(Sim + lh * 512 + 32 * kc + kbA);
            v16h gre = ld_a16(Gre + (size_t)col * 512 + 32 * kc + kbB);
            v16h gim = ld_a16(Gim + (size_t)col * 512 + 32 * kc + kbB);
            aRR = __builtin_amdgcn_wmma_f32_16x16x32_f16(false, aRe, false, gre,
                                                         (short)0, aRR, false, false);
            aII = __builtin_amdgcn_wmma_f32_16x16x32_f16(false, aIm, false, gim,
                                                         (short)0, aII, false, false);
            aRI = __builtin_amdgcn_wmma_f32_16x16x32_f16(false, aRe, false, gim,
                                                         (short)0, aRI, false, false);
            aIR = __builtin_amdgcn_wmma_f32_16x16x32_f16(false, aIm, false, gre,
                                                         (short)0, aIR, false, false);
        }
        #pragma unroll
        for (int i = 0; i < 8; ++i) {
            const int Mi = i + mO;
            const float re = aRR[i] - aII[i];
            const float im = aRI[i] + aIR[i];
            y[(size_t)(r0 + Mi) * 512 + col] = sqrtf(re * re + im * im);
        }
    }
}

// ------------------------------------------------------------------
// Wiener 2x2 helpers
// ------------------------------------------------------------------
__device__ __forceinline__ void box2x2(const float* plane, int h, int w,
                                       float& bs, float& bss) {
    float v00 = plane[h * 512 + w];
    float v01 = (w > 0)          ? plane[h * 512 + w - 1]       : 0.0f;
    float v10 = (h > 0)          ? plane[(h - 1) * 512 + w]     : 0.0f;
    float v11 = (h > 0 && w > 0) ? plane[(h - 1) * 512 + w - 1] : 0.0f;
    bs  = v00 + v01 + v10 + v11;
    bss = v00 * v00 + v01 * v01 + v10 * v10 + v11 * v11;
}

// Kernel 2: per-(b,c) mean of local variance -> noiseAcc (block reduce + atomic)
__global__ __launch_bounds__(256)
void wiener_noise_kernel(const float* __restrict__ y, float* __restrict__ noiseAcc) {
    __shared__ float red[256];
    size_t p = (size_t)blockIdx.x * 256 + threadIdx.x;
    int bc = (int)(p >> 18);            // / 262144 (block never spans bc)
    int hw = (int)(p & (PLANE - 1));
    int h = hw >> 9, w = hw & 511;
    const float* plane = y + (size_t)bc * PLANE;
    float bs, bss; box2x2(plane, h, w, bs, bss);
    float lMean = bs * 0.25f;
    float lVar  = bss * 0.25f - lMean * lMean;
    red[threadIdx.x] = lVar;
    __syncthreads();
    for (int s = 128; s > 0; s >>= 1) {
        if (threadIdx.x < s) red[threadIdx.x] += red[threadIdx.x + s];
        __syncthreads();
    }
    if (threadIdx.x == 0) atomicAdd(&noiseAcc[bc], red[0]);
}

// Kernel 3: elementwise Wiener apply
__global__ __launch_bounds__(256)
void wiener_apply_kernel(const float* __restrict__ y,
                         const float* __restrict__ noiseAcc,
                         float* __restrict__ out) {
    size_t p = (size_t)blockIdx.x * 256 + threadIdx.x;
    int bc = (int)(p >> 18);
    int hw = (int)(p & (PLANE - 1));
    int h = hw >> 9, w = hw & 511;
    const float* plane = y + (size_t)bc * PLANE;
    float bs, bss; box2x2(plane, h, w, bs, bss);
    float lMean = bs * 0.25f;
    float lVar  = bss * 0.25f - lMean * lMean;
    float noise = noiseAcc[bc] * (1.0f / (float)PLANE);
    float denom = fmaxf(fmaxf(lVar, noise), 1e-12f);
    float v   = plane[h * 512 + w];
    float res = lMean + (1.0f - noise / denom) * (v - lMean);
    out[p] = (lVar < noise) ? lMean : res;
}

// ------------------------------------------------------------------
extern "C" void kernel_launch(void* const* d_in, const int* in_sizes, int n_in,
                              void* d_out, int out_size, void* d_ws, size_t ws_size,
                              hipStream_t stream) {
    const float* img = (const float*)d_in[0];
    float* out = (float*)d_out;
    char* ws = (char*)d_ws;

    f16*   Fre      = (f16*)  (ws + OFF_FRE);
    f16*   Fim      = (f16*)  (ws + OFF_FIM);
    f16*   Gre      = (f16*)  (ws + OFF_GRE);
    f16*   Gim      = (f16*)  (ws + OFF_GIM);
    float* mw       = (float*)(ws + OFF_MW);
    float* mh       = (float*)(ws + OFF_MH);
    float* noiseAcc = (float*)(ws + OFF_NOISE);
    float* ybuf     = (float*)(ws + OFF_Y);

    init_tables<<<(512 * 512) / 256, 256, 0, stream>>>(Fre, Fim, Gre, Gim, mw, mh, noiseAcc);
    fft_filter_kernel<<<NTILES, 128, 0, stream>>>(img, Fre, Fim, Gre, Gim, mw, mh, ybuf);
    wiener_noise_kernel<<<(unsigned)(NPIX / 256), 256, 0, stream>>>(ybuf, noiseAcc);
    wiener_apply_kernel<<<(unsigned)(NPIX / 256), 256, 0, stream>>>(ybuf, noiseAcc, out);
}
